// SVHNCapsuleNet_46961172415010
// MI455X (gfx1250) — matrix operand order repaired and checked by reading, not compiled, not run on hardware
//
#include <hip/hip_runtime.h>
#include <hip/hip_fp16.h>

typedef __attribute__((ext_vector_type(16))) _Float16 v16h;
typedef __attribute__((ext_vector_type(8)))  _Float16 v8h;
typedef __attribute__((ext_vector_type(8)))  float    v8f;

#define EPS_SQ 1e-12f

union Frag16 { v16h v; v8h h[2]; };

// ---------------------------------------------------------------------------
// Activation -> f16 load helpers (f32 input layer, f16 intermediate layers)
// ---------------------------------------------------------------------------
__device__ __forceinline__ __half load_h(const float* p)  { return __float2half(*p); }
__device__ __forceinline__ __half load_h(const __half* p) { return *p; }

// XOR-butterfly add via ds_swizzle_b32 (group-of-32: and=0x1f, or=0, xor=MASK).
// Single LDS-pipe op, immediate-encoded pattern, no address VGPR setup.
template <int MASK>
__device__ __forceinline__ float xor_add(float s)
{
    int r = __builtin_amdgcn_ds_swizzle(__float_as_int(s), 0x7C00 | MASK);
    return s + __int_as_float(r);
}

// ---------------------------------------------------------------------------
// Pre-pack conv weights (OIHW fp32) into WMMA B-fragment order, f16.
// Layout: [ntile][kchunk][lane(32)][16 f16]  ; lane: N = lane&15, Khalf = lane>>4
// B fragment element j (0..15): K = kc*32 + j + 16*Khalf
// ---------------------------------------------------------------------------
__global__ void pack_weights_kernel(const float* __restrict__ w,
                                    __half* __restrict__ pack,
                                    int ktot, int kc_cnt, int total)
{
    int idx = blockIdx.x * 256 + threadIdx.x;
    if (idx >= total) return;
    int j    = idx & 15;
    int lane = (idx >> 4) & 31;
    int kc   = (idx >> 9) % kc_cnt;
    int nt   = idx / (kc_cnt * 512);
    int half = lane >> 4;
    int n    = nt * 16 + (lane & 15);
    int kk   = kc * 32 + j + 16 * half;
    float v  = (kk < ktot) ? w[(size_t)n * ktot + kk] : 0.0f;
    pack[idx] = __float2half(v);
}

// ---------------------------------------------------------------------------
// LDS-staged implicit-GEMM capsule conv + fused squash epilogue.
//  - Block of 256 threads stages an im2col tile (MB pixels x K_pad, f16) into
//    LDS with coalesced global loads.
//  - Each wave owns one 16(pixel) x 16(channel) tile; A fragments come from
//    LDS via two ds_load_b128 per K-chunk, B from the pre-packed global image.
//  - K loop fully unrolled with TWO accumulators (even/odd chunks) so the
//    WMMA->WMMA D->C dependency distance doubles and the matrix pipe doesn't
//    eat hazard NOPs between consecutive v_wmma_f32_16x16x32_f16.
// ---------------------------------------------------------------------------
template <typename AT, int CIN, int H, int W, int COUT, int HOUT, int WOUT,
          int STRIDE, int PAD, int KH, int KW, int OL>
__global__ __launch_bounds__(256) void caps_conv_kernel(
    const AT* __restrict__ x, const __half* __restrict__ packB,
    __half* __restrict__ y)
{
    constexpr int KTOT  = CIN * KH * KW;
    constexpr int KC    = (KTOT + 31) / 32;
    constexpr int KPAD  = KC * 32;
    constexpr int NT    = COUT / 16;
    constexpr int MB    = 128 / NT;          // pixels per block (8 waves = MB/16 * NT tiles)
    constexpr int PITCH = KPAD + 8;          // halves; +8 => 16B-aligned rows, bank skew
    constexpr int PIX   = HOUT * WOUT;

    __shared__ __half sIm[MB * PITCH];

    const int t        = threadIdx.x;
    const int block_m0 = blockIdx.x * MB;

    // ---- cooperative im2col staging (K-major: consecutive threads ->
    //      consecutive pixels -> coalesced global addresses) ----------------
    for (int idx = t; idx < MB * KPAD; idx += 256) {
        int mr = idx % MB;
        int kk = idx / MB;
        __half hv = __half(0.0f);
        if (kk < KTOT) {
            int m   = block_m0 + mr;
            int b   = m / PIX;
            int pix = m % PIX;
            int oy  = pix / WOUT, ox = pix % WOUT;
            int ci  = kk / (KH * KW);
            int r   = kk % (KH * KW);
            int ky  = r / KW, kx = r % KW;
            int iy  = oy * STRIDE - PAD + ky;
            int ix  = ox * STRIDE - PAD + kx;
            if (iy >= 0 && iy < H && ix >= 0 && ix < W)
                hv = load_h(x + ((size_t)(b * CIN + ci) * H + iy) * W + ix);
        }
        sIm[mr * PITCH + kk] = hv;
    }
    __syncthreads();

    // ---- wave-level WMMA tiles -------------------------------------------
    const int wave = t >> 5;
    const int lane = t & 31;
    const int half = lane >> 4;       // K-half (A/B), M-half (C)
    const int mrow = lane & 15;       // A row / C column

    const int nt  = wave % NT;        // channel tile
    const int lmt = wave / NT;        // local pixel tile

    const __half* srow = &sIm[(lmt * 16 + mrow) * PITCH];
    const __half* bp   = packB + (size_t)nt * KC * 512 + lane * 16;

    v8f acc0 = {};
    v8f acc1 = {};
#pragma unroll
    for (int kc = 0; kc < KC; ++kc) {
        // A fragment: j=0..7  -> K = kc*32 + 16*half + 0..7      (contiguous)
        //             j=8..15 -> K = kc*32 + 16*half + 8 + 0..7  (contiguous)
        Frag16 a;
        a.h[0] = *reinterpret_cast<const v8h*>(srow + kc * 32 + 16 * half);
        a.h[1] = *reinterpret_cast<const v8h*>(srow + kc * 32 + 16 * half + 8);
        v16h bfrag = *reinterpret_cast<const v16h*>(bp + (size_t)kc * 512);
        if (kc & 1)
            acc1 = __builtin_amdgcn_wmma_f32_16x16x32_f16(
                false, a.v, false, bfrag, (short)0, acc1, false, false);
        else
            acc0 = __builtin_amdgcn_wmma_f32_16x16x32_f16(
                false, a.v, false, bfrag, (short)0, acc0, false, false);
    }
    v8f acc = acc0 + acc1;

    // ---- fused squash: per C row (pixel), |s|^2 over OL channels ----------
    // Channels live across the 16-lane half (N = lane&15); OL in {4,8,16} =>
    // the xor butterfly (masks < 16) stays inside the half, groups align.
    const int cch = nt * 16 + (lane & 15);
#pragma unroll
    for (int r = 0; r < 8; ++r) {
        float v  = acc[r];
        float sq = v * v;
        sq = xor_add<1>(sq);
        sq = xor_add<2>(sq);
        if constexpr (OL >= 8)  sq = xor_add<4>(sq);
        if constexpr (OL >= 16) sq = xor_add<8>(sq);
        float scale = (sq / (1.0f + sq)) * rsqrtf(sq + EPS_SQ);

        int M  = block_m0 + lmt * 16 + r + 8 * half;   // C layout: row = vgpr + 8*half
        int bb = M / PIX;
        int pp = M % PIX;
        y[((size_t)(bb * COUT + cch)) * PIX + pp] = __float2half(v * scale);
    }
}

// ---------------------------------------------------------------------------
// Capsule linear + 3-iteration dynamic routing + class norms.
// One block per batch element; everything LDS-resident. Both uses of the
// priors factor through the 16-dim l axis, so priors are never materialized.
// u[b,i,l] = conv6[b, (i&3)*16 + l, i>>2]   (reshape/swapaxes algebra)
// ---------------------------------------------------------------------------
__global__ __launch_bounds__(256) void routing_kernel(
    const __half* __restrict__ act6, const float* __restrict__ wc,
    float* __restrict__ out)
{
    __shared__ float sU[64][16];        // input capsules
    __shared__ float sWC[10][32][16];   // wc[o][d][l]
    __shared__ float sLog[10][64];      // routing logits
    __shared__ float sProb[10][64];
    __shared__ float sS[10][16];        // s[o][l] = sum_i prob*u
    __shared__ float sOut[10][32];      // v[o][d]
    __shared__ float sT[10][16];        // t[o][l] = sum_d wc*v

    const int b = blockIdx.x;
    const int t = threadIdx.x;

    for (int idx = t; idx < 64 * 16; idx += 256) {
        int i = idx >> 4, l = idx & 15;
        int c = (i & 3) * 16 + l;
        int s = i >> 2;
        sU[i][l] = __half2float(act6[((size_t)(b * 64 + c)) * 16 + s]);
    }
    for (int idx = t; idx < 10 * 32 * 16; idx += 256)
        (&sWC[0][0][0])[idx] = wc[idx];
    for (int idx = t; idx < 640; idx += 256)
        (&sLog[0][0])[idx] = 0.0f;
    __syncthreads();

    for (int it = 0; it < 3; ++it) {
        // softmax over output capsules (axis o), per input capsule i
        if (t < 64) {
            int i = t;
            float mx = -1e30f;
            for (int o = 0; o < 10; ++o) mx = fmaxf(mx, sLog[o][i]);
            float e[10], sum = 0.0f;
            for (int o = 0; o < 10; ++o) { e[o] = __expf(sLog[o][i] - mx); sum += e[o]; }
            float inv = 1.0f / sum;
            for (int o = 0; o < 10; ++o) sProb[o][i] = e[o] * inv;
        }
        __syncthreads();
        // s[o][l] = sum_i prob[o][i] * u[i][l]
        if (t < 160) {
            int o = t >> 4, l = t & 15;
            float a = 0.0f;
            for (int i = 0; i < 64; ++i) a += sProb[o][i] * sU[i][l];
            sS[o][l] = a;
        }
        __syncthreads();
        // v[o][d] = sum_l wc[o][d][l] * s[o][l]
        for (int idx = t; idx < 320; idx += 256) {
            int o = idx >> 5, d = idx & 31;
            float a = 0.0f;
            for (int l = 0; l < 16; ++l) a += sWC[o][d][l] * sS[o][l];
            sOut[o][d] = a;
        }
        __syncthreads();
        // squash over d
        if (t < 10) {
            float sq = 0.0f;
            for (int d = 0; d < 32; ++d) sq += sOut[t][d] * sOut[t][d];
            float scale = (sq / (1.0f + sq)) * rsqrtf(sq + EPS_SQ);
            for (int d = 0; d < 32; ++d) sOut[t][d] *= scale;
        }
        __syncthreads();
        if (it < 2) {
            // t[o][l] = sum_d wc[o][d][l] * v[o][d]
            if (t < 160) {
                int o = t >> 4, l = t & 15;
                float a = 0.0f;
                for (int d = 0; d < 32; ++d) a += sWC[o][d][l] * sOut[o][d];
                sT[o][l] = a;
            }
            __syncthreads();
            // logits[o][i] += sum_l u[i][l] * t[o][l]
            for (int idx = t; idx < 640; idx += 256) {
                int o = idx / 64, i = idx % 64;
                float a = 0.0f;
                for (int l = 0; l < 16; ++l) a += sU[i][l] * sT[o][l];
                sLog[o][i] += a;
            }
            __syncthreads();
        }
    }
    if (t < 10) {
        float sq = 0.0f;
        for (int d = 0; d < 32; ++d) sq += sOut[t][d] * sOut[t][d];
        out[(size_t)b * 10 + t] = sqrtf(sq + EPS_SQ);
    }
}

// ---------------------------------------------------------------------------
// Host launch
// ---------------------------------------------------------------------------
extern "C" void kernel_launch(void* const* d_in, const int* in_sizes, int n_in,
                              void* d_out, int out_size, void* d_ws, size_t ws_size,
                              hipStream_t stream)
{
    (void)in_sizes; (void)n_in; (void)out_size; (void)ws_size;

    const float* x  = (const float*)d_in[0];
    const float* w1 = (const float*)d_in[1];
    const float* w2 = (const float*)d_in[2];
    const float* w3 = (const float*)d_in[3];
    const float* w4 = (const float*)d_in[4];
    const float* w5 = (const float*)d_in[5];
    const float* w6 = (const float*)d_in[6];
    const float* wc = (const float*)d_in[7];
    float* out = (float*)d_out;

    const int B = 1024;
    char* ws = (char*)d_ws;

    // Workspace layout (bytes). Packs <205KB, f16 act ping-pong ~40MB.
    __half* p1 = (__half*)(ws + 0);        // NT=1 KC=3  ->  3,072 B
    __half* p2 = (__half*)(ws + 8192);     // NT=1 KC=5  ->  5,120 B
    __half* p3 = (__half*)(ws + 16384);    // NT=2 KC=5  -> 10,240 B
    __half* p4 = (__half*)(ws + 32768);    // NT=2 KC=9  -> 18,432 B
    __half* p5 = (__half*)(ws + 65536);    // NT=4 KC=9  -> 36,864 B
    __half* p6 = (__half*)(ws + 131072);   // NT=4 KC=18 -> 73,728 B
    __half* bufA = (__half*)(ws + 262144);            // 33,554,432 B max
    __half* bufB = (__half*)(ws + 262144 + 33554432); //  8,388,608 B max

    // --- pack weights into B-fragment order -------------------------------
    struct PackDesc { const float* w; __half* p; int ktot, kc, nt; };
    const PackDesc packs[6] = {
        { w1, p1,  3 * 25, 3, 1 },
        { w2, p2, 16 * 9,  5, 1 },
        { w3, p3, 16 * 9,  5, 2 },
        { w4, p4, 32 * 9,  9, 2 },
        { w5, p5, 32 * 9,  9, 4 },
        { w6, p6, 64 * 9, 18, 4 },
    };
    for (int i = 0; i < 6; ++i) {
        int total = packs[i].nt * packs[i].kc * 512;
        pack_weights_kernel<<<(total + 255) / 256, 256, 0, stream>>>(
            packs[i].w, packs[i].p, packs[i].ktot, packs[i].kc, total);
    }

    // --- conv + squash chain (ping-pong f16 activations) ------------------
    // blocks = total_pixels / MB  (MB = 128/NT); all divisions are exact.
    {   // L1: [B,3,32,32] -> [B,16,32,32], s1 p2 k5, OL=4  (MB=128)
        caps_conv_kernel<float, 3, 32, 32, 16, 32, 32, 1, 2, 5, 5, 4>
            <<<B * 32 * 32 / 128, 256, 0, stream>>>(x, p1, bufA);
    }
    {   // L2: -> [B,16,16,16], s2 p1 k3, OL=4  (MB=128)
        caps_conv_kernel<__half, 16, 32, 32, 16, 16, 16, 2, 1, 3, 3, 4>
            <<<B * 16 * 16 / 128, 256, 0, stream>>>(bufA, p2, bufB);
    }
    {   // L3: -> [B,32,16,16], s1 p1 k3, OL=8  (MB=64)
        caps_conv_kernel<__half, 16, 16, 16, 32, 16, 16, 1, 1, 3, 3, 8>
            <<<B * 16 * 16 / 64, 256, 0, stream>>>(bufB, p3, bufA);
    }
    {   // L4: -> [B,32,8,8], s2 p1 k3, OL=8  (MB=64)
        caps_conv_kernel<__half, 32, 16, 16, 32, 8, 8, 2, 1, 3, 3, 8>
            <<<B * 8 * 8 / 64, 256, 0, stream>>>(bufA, p4, bufB);
    }
    {   // L5: -> [B,64,8,8], s1 p1 k3, OL=16  (MB=32)
        caps_conv_kernel<__half, 32, 8, 8, 64, 8, 8, 1, 1, 3, 3, 16>
            <<<B * 8 * 8 / 32, 256, 0, stream>>>(bufB, p5, bufA);
    }
    {   // L6: -> [B,64,4,4], s2 p1 k3, OL=16  (MB=32)
        caps_conv_kernel<__half, 64, 8, 8, 64, 4, 4, 2, 1, 3, 3, 16>
            <<<B * 4 * 4 / 32, 256, 0, stream>>>(bufA, p6, bufB);
    }

    // --- capsule linear + dynamic routing + class norms -------------------
    routing_kernel<<<B, 256, 0, stream>>>(bufB, wc, out);
}
